// AxialAttention_28269474743064
// MI455X (gfx1250) — compile-verified, matrix operand order checked by reading, not run on hardware
//
#include <hip/hip_runtime.h>
#include <hip/hip_bf16.h>

// ---------------------------------------------------------------------------
// Axial attention for MI455X (gfx1250, wave32, WMMA + async global->LDS).
// All matmuls use v_wmma_f32_16x16x32_bf16 (16x16 tiles, K=32 per step).
// ---------------------------------------------------------------------------

typedef __attribute__((ext_vector_type(16))) __bf16 v16bf;
typedef __attribute__((ext_vector_type(8)))  float  v8f;

#define NB      8
#define NC      256
#define NH      128
#define NW      128
#define NHEADS  8
#define HD      32
#define HWSZ    (NH * NW)        // 16384
#define NPOS    128              // attention length (both axes)
#define SCALE   0.17677669529663687f  // 1/sqrt(32)

__device__ __forceinline__ v8f wmma_bf16(v16bf a, v16bf b, v8f c) {
  // (neg_a, A, neg_b, B, c_mod, C, reuse_a, reuse_b)
  return __builtin_amdgcn_wmma_f32_16x16x32_bf16(false, a, false, b, (short)0, c,
                                                 false, false);
}

// Async global->LDS copy, 16 bytes per call per lane (ASYNCcnt tracked).
// lds_off: LDS byte address (low 32 bits of flat shared address per ISA
// aperture rule).
__device__ __forceinline__ void async_ld_b128(unsigned lds_off, const void* g) {
  asm volatile("global_load_async_to_lds_b128 %0, %1, off"
               :: "v"(lds_off), "v"((unsigned long long)(uintptr_t)g)
               : "memory");
}
__device__ __forceinline__ void wait_async0() {
  asm volatile("s_wait_asynccnt 0x0" ::: "memory");
}
__device__ __forceinline__ unsigned lds_off32(const void* p) {
  return (unsigned)(uintptr_t)p;
}

// ---------------------------------------------------------------------------
// Kernel 1: x [B,C,H,W] f32  ->  xT [B, H*W, C] bf16   (LDS-tiled transpose)
// ---------------------------------------------------------------------------
__global__ __launch_bounds__(256) void k_transpose_cast(
    const float* __restrict__ x, __bf16* __restrict__ xT) {
  __shared__ float tile[32][33];
  int b   = blockIdx.z;
  int hw0 = blockIdx.x * 32;
  int c0  = blockIdx.y * 32;
  const float* xb = x + (size_t)b * NC * HWSZ;
  for (int i = threadIdx.y; i < 32; i += 8)
    tile[i][threadIdx.x] = xb[(size_t)(c0 + i) * HWSZ + hw0 + threadIdx.x];
  __syncthreads();
  __bf16* ob = xT + (size_t)b * HWSZ * NC;
  for (int i = threadIdx.y; i < 32; i += 8)
    ob[(size_t)(hw0 + i) * NC + c0 + threadIdx.x] = (__bf16)tile[threadIdx.x][i];
}

// ---------------------------------------------------------------------------
// Kernel 2: cast all four weight matrices to bf16, layout kept [C_out][C_in]
// (that is exactly the per-lane contiguous B-fragment layout).
// ---------------------------------------------------------------------------
__global__ __launch_bounds__(256) void k_cast_weights(
    const float* __restrict__ wq, const float* __restrict__ wk,
    const float* __restrict__ wv, const float* __restrict__ wo,
    __bf16* __restrict__ dq, __bf16* __restrict__ dk,
    __bf16* __restrict__ dv, __bf16* __restrict__ dwo) {
  int i = blockIdx.x * blockDim.x + threadIdx.x;
  if (i < NC * NC) {
    dq[i]  = (__bf16)wq[i];
    dk[i]  = (__bf16)wk[i];
    dv[i]  = (__bf16)wv[i];
    dwo[i] = (__bf16)wo[i];
  }
}

// ---------------------------------------------------------------------------
// Kernel 3: QKV projection GEMM.  Y[bhw, 3*C] = X[bhw, C] * W^T  (+bias)
// Block: 256 threads = 8 waves. Block stages one 16-row strip of X into LDS
// via async global->LDS; each wave computes one 16x16 output tile (8 WMMA
// k-steps) and scatters bf16 results into q/k/v[b][head][y][w][d].
// ---------------------------------------------------------------------------
__global__ __launch_bounds__(256) void k_qkv_gemm(
    const __bf16* __restrict__ xT,
    const __bf16* __restrict__ wq, const __bf16* __restrict__ wk,
    const __bf16* __restrict__ wv,
    const float* __restrict__ bq, const float* __restrict__ bk,
    const float* __restrict__ bv,
    __bf16* __restrict__ qb, __bf16* __restrict__ kb, __bf16* __restrict__ vb) {
  __shared__ alignas(32) __bf16 as[16 * NC];   // 8 KB A strip
  int rowtile = blockIdx.x;
  int tid = threadIdx.x;
  {
    int row = tid >> 4, ch = tid & 15;
    unsigned loff = lds_off32(as + row * NC + ch * 16);
    const __bf16* g = xT + (size_t)(rowtile * 16 + row) * NC + ch * 16;
    async_ld_b128(loff, g);
    async_ld_b128(loff + 16, g + 8);
  }
  wait_async0();
  __syncthreads();

  int wave = tid >> 5, lane = tid & 31, half = lane >> 4, l15 = lane & 15;
  int t   = blockIdx.y * 8 + wave;   // 0..47 : which of the 48 column tiles
  int mat = t >> 4, n0 = (t & 15) * 16;
  const __bf16* wm = (mat == 0) ? wq : (mat == 1) ? wk : wv;
  const float*  bm = (mat == 0) ? bq : (mat == 1) ? bk : bv;
  __bf16*       dst = (mat == 0) ? qb : (mat == 1) ? kb : vb;

  v8f acc = {};
  for (int s = 0; s < 8; ++s) {
    if (s < 7)
      __builtin_prefetch(wm + (size_t)(n0 + l15) * NC + (s + 1) * 32 + half * 16, 0, 0);
    v16bf a = *(const v16bf*)(as + l15 * NC + s * 32 + half * 16);
    v16bf b = *(const v16bf*)(wm + (size_t)(n0 + l15) * NC + s * 32 + half * 16);
    acc = wmma_bf16(a, b, acc);
  }
  float bias = bm[n0 + l15];
  int cc = n0 + l15, head = cc >> 5, d = cc & (HD - 1);
  for (int j = 0; j < 8; ++j) {
    int r  = rowtile * 16 + j + 8 * half;
    int b_ = r >> 14, rem = r & (HWSZ - 1), y = rem >> 7, w = rem & (NW - 1);
    dst[((((size_t)b_ * NHEADS + head) * NH + y) * NW + w) * HD + d] =
        (__bf16)(acc[j] + bias);
  }
}

// ---------------------------------------------------------------------------
// Kernels 4/5: axial attention. One workgroup per (b, head, line).
// VERT=0: line = row y, attend over w.  VERT=1: line = col w, attend over y.
// K block staged once per workgroup via async global->LDS (kills the 8x
// redundant per-wave K reads); V^T staged for the PV GEMM B-operand.
// 8 waves; wave owns a 16x128 score strip. Softmax fully in registers using
// wave32 shfl_xor over the 16-lane halves of the C/D tile layout.
// ---------------------------------------------------------------------------
template <int VERT>
__global__ __launch_bounds__(256) void k_attn(
    const __bf16* __restrict__ qb, const __bf16* __restrict__ kb,
    const __bf16* __restrict__ vb, __bf16* __restrict__ ob) {
  __shared__ alignas(32) __bf16 sm_attn[NPOS * NPOS];  // 32 KB probabilities
  __shared__ alignas(32) __bf16 sm_k[NPOS * HD];       //  8 KB K block
  __shared__ alignas(32) __bf16 sm_vT[HD * NPOS];      //  8 KB V transposed

  int idx   = blockIdx.x;
  int fixed = idx & 127;
  int head  = (idx >> 7) & 7;
  int b     = idx >> 10;
  size_t base = (((size_t)b * NHEADS + head) * NH) * NW * HD;
  size_t org  = VERT ? base + (size_t)fixed * HD : base + (size_t)fixed * NW * HD;
  size_t rstr = VERT ? (size_t)NW * HD : (size_t)HD;   // stride between positions

  int tid = threadIdx.x, wave = tid >> 5, lane = tid & 31;
  int half = lane >> 4, l15 = lane & 15;
  int m0 = wave * 16;

  // stage K block: sm_k[p][d] = K[p][d]  (async global->LDS, 2x b128/thread)
  {
    int p = tid >> 1, ch = tid & 1;
    unsigned loff = lds_off32(sm_k + p * HD + ch * 16);
    const __bf16* g = kb + org + (size_t)p * rstr + ch * 16;
    async_ld_b128(loff, g);
    async_ld_b128(loff + 16, g + 8);
  }
  // stage V^T: sm_vT[d][p] = V[p][d]
  for (int i = tid; i < HD * NPOS; i += 256) {
    int d = i & (HD - 1), p = i >> 5;
    sm_vT[d * NPOS + p] = vb[org + (size_t)p * rstr + d];
  }
  wait_async0();
  __syncthreads();

  // S = Q K^T  (K-dim = HD = 32 -> single WMMA per 16x16 tile)
  v16bf qf = *(const v16bf*)(qb + org + (size_t)(m0 + l15) * rstr + half * 16);
  v8f acc[8];
  for (int t = 0; t < 8; ++t) {
    v16bf kf = *(const v16bf*)(sm_k + (t * 16 + l15) * HD + half * 16);
    v8f z = {};
    acc[t] = wmma_bf16(qf, kf, z);
  }

  // In-register softmax. Row (j + 8*half) lives in VGPR j across lanes of one
  // 16-lane half; xor masks 1,2,4,8 never cross lane bit 4.
  for (int j = 0; j < 8; ++j) {
    float m = -3.0e38f;
    for (int t = 0; t < 8; ++t) {
      acc[t][j] = acc[t][j] * SCALE;
      m = fmaxf(m, acc[t][j]);
    }
    for (int off = 1; off < 16; off <<= 1) m = fmaxf(m, __shfl_xor(m, off, 32));
    float s = 0.f;
    for (int t = 0; t < 8; ++t) {
      float e = __expf(acc[t][j] - m);
      acc[t][j] = e;
      s += e;
    }
    for (int off = 1; off < 16; off <<= 1) s += __shfl_xor(s, off, 32);
    float inv = 1.0f / s;
    int row = m0 + j + 8 * half;
    for (int t = 0; t < 8; ++t)
      sm_attn[row * NPOS + t * 16 + l15] = (__bf16)(acc[t][j] * inv);
  }
  __syncthreads();

  // O = P V   (K = 128 positions -> 4 k-steps; N = 32 -> 2 tiles)
  for (int n = 0; n < 2; ++n) {
    v8f o = {};
    for (int kk = 0; kk < 4; ++kk) {
      v16bf a  = *(const v16bf*)(sm_attn + (m0 + l15) * NPOS + kk * 32 + half * 16);
      v16bf vv = *(const v16bf*)(sm_vT  + (n * 16 + l15) * NPOS + kk * 32 + half * 16);
      o = wmma_bf16(a, vv, o);
    }
    for (int j = 0; j < 8; ++j) {
      int p = m0 + j + 8 * half;
      ob[org + (size_t)p * rstr + n * 16 + l15] = (__bf16)o[j];
    }
  }
}

// ---------------------------------------------------------------------------
// Kernel 6: output projection  out = (oh + ow) * Wo^T + bo + x
// Block: one 16-row strip; stage bf16(oh+ow) in LDS; 8 waves x 2 tiles cover
// all 256 output channels; fused bias + residual; fp32 [B,C,H,W] store.
// ---------------------------------------------------------------------------
__global__ __launch_bounds__(256) void k_out_proj(
    const __bf16* __restrict__ oh, const __bf16* __restrict__ ow,
    const __bf16* __restrict__ wo, const float* __restrict__ bo,
    const float* __restrict__ x, float* __restrict__ out) {
  __shared__ alignas(32) __bf16 asum[16 * NC];   // 8 KB
  int rowtile = blockIdx.x;
  int tid = threadIdx.x;
  {
    int row = tid >> 4, ch = tid & 15;
    int r  = rowtile * 16 + row;
    int b_ = r >> 14, rem = r & (HWSZ - 1), y = rem >> 7, w = rem & (NW - 1);
    int head = ch >> 1, d0 = (ch & 1) * 16;
    size_t src = ((((size_t)b_ * NHEADS + head) * NH + y) * NW + w) * HD + d0;
    v16bf a = *(const v16bf*)(oh + src);
    v16bf c = *(const v16bf*)(ow + src);
    __bf16* dstp = asum + row * NC + ch * 16;
    for (int i = 0; i < 16; ++i) dstp[i] = (__bf16)((float)a[i] + (float)c[i]);
  }
  __syncthreads();

  int wave = tid >> 5, lane = tid & 31, half = lane >> 4, l15 = lane & 15;
  for (int nt = wave * 2; nt < wave * 2 + 2; ++nt) {
    int n0 = nt * 16;
    v8f acc = {};
    for (int s = 0; s < 8; ++s) {
      v16bf a = *(const v16bf*)(asum + l15 * NC + s * 32 + half * 16);
      v16bf b = *(const v16bf*)(wo + (size_t)(n0 + l15) * NC + s * 32 + half * 16);
      acc = wmma_bf16(a, b, acc);
    }
    int cc = n0 + l15;
    float bias = bo[cc];
    for (int j = 0; j < 8; ++j) {
      int r  = rowtile * 16 + j + 8 * half;
      int b_ = r >> 14, rem = r & (HWSZ - 1), y = rem >> 7, w = rem & (NW - 1);
      size_t xi = (((size_t)b_ * NC + cc) * NH + y) * NW + w;
      out[xi] = acc[j] + bias + x[xi];
    }
  }
}

// ---------------------------------------------------------------------------
// Host-side orchestration (graph-capture safe: only kernel launches).
// ---------------------------------------------------------------------------
extern "C" void kernel_launch(void* const* d_in, const int* in_sizes, int n_in,
                              void* d_out, int out_size, void* d_ws, size_t ws_size,
                              hipStream_t stream) {
  const float* x  = (const float*)d_in[0];
  const float* Wq = (const float*)d_in[1];
  const float* bq = (const float*)d_in[2];
  const float* Wk = (const float*)d_in[3];
  const float* bk = (const float*)d_in[4];
  const float* Wv = (const float*)d_in[5];
  const float* bv = (const float*)d_in[6];
  const float* Wo = (const float*)d_in[7];
  const float* bo = (const float*)d_in[8];
  float* out = (float*)d_out;

  char* ws = (char*)d_ws;
  const size_t SZT = (size_t)NB * HWSZ * NC * 2;  // 64 MiB per bf16 tensor
  __bf16* xT  = (__bf16*)(ws);
  __bf16* qb  = (__bf16*)(ws + 1 * SZT);
  __bf16* kb  = (__bf16*)(ws + 2 * SZT);
  __bf16* vb  = (__bf16*)(ws + 3 * SZT);
  __bf16* ohb = (__bf16*)(ws + 4 * SZT);
  __bf16* owb = (__bf16*)(ws + 5 * SZT);
  __bf16* wqb = (__bf16*)(ws + 6 * SZT);
  __bf16* wkb = wqb + NC * NC;
  __bf16* wvb = wkb + NC * NC;
  __bf16* wob = wvb + NC * NC;

  // 1) transpose + cast x
  dim3 tb(32, 8);
  dim3 tg(HWSZ / 32, NC / 32, NB);
  k_transpose_cast<<<tg, tb, 0, stream>>>(x, xT);

  // 2) cast weights
  k_cast_weights<<<(NC * NC + 255) / 256, 256, 0, stream>>>(
      Wq, Wk, Wv, Wo, wqb, wkb, wvb, wob);

  // 3) QKV projection
  dim3 gq(NB * HWSZ / 16, 6);  // 8192 row tiles x (48 col tiles / 8 waves)
  k_qkv_gemm<<<gq, 256, 0, stream>>>(xT, wqb, wkb, wvb, bq, bk, bv, qb, kb, vb);

  // 4) horizontal attention, 5) vertical attention
  k_attn<0><<<NB * NHEADS * NH, 256, 0, stream>>>(qb, kb, vb, ohb);
  k_attn<1><<<NB * NHEADS * NW, 256, 0, stream>>>(qb, kb, vb, owb);

  // 6) output projection + bias + residual
  k_out_proj<<<NB * HWSZ / 16, 256, 0, stream>>>(ohb, owb, wob, bo, x, out);
}
